// GeoLlamaHybrid_50878182589110
// MI455X (gfx1250) — matrix operand order repaired and checked by reading, not compiled, not run on hardware
//
#include <hip/hip_runtime.h>
#include <hip/hip_bf16.h>
#include <cstdint>
#include <cstddef>

// ---------------------------------------------------------------------------
// GeoLlamaHybrid for MI455X (gfx1250, wave32)
// Phase 0: W_lift f32 [k][n] -> Wt bf16 [n][k]  (transposed, 8 MB, L2-resident)
// Phase 1: rotors = X @ W_lift + b  via v_wmma_f32_16x16x32_bf16
//          B tile double-buffered via GLOBAL_LOAD_ASYNC_TO_LDS_B128:
//          next tile's copy stays in flight under the current tile's WMMAs
// Phase 2: sequential rotor scan, one wave per two heads (h, h+32), gp via cross-lane shfl
// ---------------------------------------------------------------------------

typedef __attribute__((ext_vector_type(16))) __bf16 v16bf;
typedef __attribute__((ext_vector_type(8)))  float  v8f;

#define D_MODEL 2048
#define SEQ     4096
#define BSZ     4
#define NHEADS  64
#define MTOT    (BSZ * SEQ)      // 16384
#define EPSV    1e-8f

// GEMM tiling: 128x128 block tile, K-step 32, 8 waves (2M x 4N), 64x32 per wave
#define BM 128
#define BN 128
#define BK 32
#define LDSA 40                  // padded row stride (bf16): 80 B -> 16B-aligned chunks,
#define LDSB 40                  // conflict-free fragment reads

// ---- CDNA5 async global->LDS copy (ASYNCcnt-tracked), per CDNA5 ISA ----
__device__ __forceinline__ void async_copy_b128(void* lds_dst, const void* gsrc) {
  uint32_t loff = (uint32_t)(uintptr_t)lds_dst;     // LDS aperture: addr[31:0] = offset
  uint64_t ga   = (uint64_t)(uintptr_t)gsrc;
  asm volatile("global_load_async_to_lds_b128 %0, %1, off"
               :: "v"(loff), "v"(ga) : "memory");
}
__device__ __forceinline__ void wait_asynccnt0() {
#if __has_builtin(__builtin_amdgcn_s_wait_asynccnt)
  __builtin_amdgcn_s_wait_asynccnt(0);
#else
  asm volatile("s_wait_asynccnt 0x0" ::: "memory");
#endif
}
__device__ __forceinline__ void wait_asynccnt2() {
#if __has_builtin(__builtin_amdgcn_s_wait_asynccnt)
  __builtin_amdgcn_s_wait_asynccnt(2);
#else
  asm volatile("s_wait_asynccnt 0x2" ::: "memory");
#endif
}

// ---------------------------------------------------------------------------
// Wt[n][k] = (bf16) W[k][n]
__global__ __launch_bounds__(256) void cvt_wT_kernel(const float* __restrict__ w,
                                                     __bf16* __restrict__ wt) {
  int idx = blockIdx.x * 256 + threadIdx.x;          // 4M threads
  int n = idx >> 11, k = idx & 2047;
  wt[idx] = (__bf16)w[(size_t)k * D_MODEL + n];
}

// ---------------------------------------------------------------------------
// rotors[m][n] = sum_k X[m][k] * W[k][n] + bias[n]   (m = b*SEQ+s, n = h*32+c)
__global__ __launch_bounds__(256) void lift_gemm_kernel(
    const float*  __restrict__ X,     // [MTOT, D_MODEL] f32
    const __bf16* __restrict__ Wt,    // [D_MODEL(n), D_MODEL(k)] bf16 (transposed)
    const float*  __restrict__ bias,  // [D_MODEL] f32
    __bf16*       __restrict__ out)   // [MTOT, D_MODEL] bf16
{
  __shared__ __align__(16) __bf16 As[BM * LDSA];     // [m][k]
  __shared__ __align__(16) __bf16 Bs[2][BN * LDSB];  // [n][k], double buffered

  const int tid  = threadIdx.x;
  const int lane = tid & 31;
  const int wid  = tid >> 5;
  const int wm   = wid >> 2;       // 0..1  -> 64-row slab
  const int wn   = wid & 3;        // 0..3  -> 32-col slab
  const int half = lane >> 4;      // lane half (ISA A/B fragment layout)
  const int l16  = lane & 15;

  // N fastest in the grid: A panel (1 MB) reused from L2 across N-blocks,
  // Wt (8 MB bf16) fully L2-resident.
  const int n0 = blockIdx.x * BN;
  const int m0 = blockIdx.y * BM;

  // per-thread chunk of the B tile copy: 512 x 16B chunks / 256 threads
  const int bn0 = tid >> 2, bj0 = (tid & 3) * 8;
  const int bn1 = (256 + tid) >> 2, bj1 = ((256 + tid) & 3) * 8;

  v8f acc[4][2] = {};

  // prologue: async B copy for tile 0
  async_copy_b128(&Bs[0][bn0 * LDSB + bj0], Wt + (size_t)(n0 + bn0) * D_MODEL + bj0);
  async_copy_b128(&Bs[0][bn1 * LDSB + bj1], Wt + (size_t)(n0 + bn1) * D_MODEL + bj1);

  const int NT = D_MODEL / BK;     // 64
  for (int it = 0; it < NT; ++it) {
    const int kt = it * BK;
    __syncthreads();   // close previous iteration's fragment reads

    // issue async B copy for the NEXT tile; it stays in flight under the WMMAs
    if (it + 1 < NT) {
      const int ktn = kt + BK;
      __bf16* bd = Bs[(it + 1) & 1];
      async_copy_b128(&bd[bn0 * LDSB + bj0], Wt + (size_t)(n0 + bn0) * D_MODEL + ktn + bj0);
      async_copy_b128(&bd[bn1 * LDSB + bj1], Wt + (size_t)(n0 + bn1) * D_MODEL + ktn + bj1);
    }

    // A tile: 128x32 f32 -> bf16 (overlaps with in-flight async copies)
    #pragma unroll
    for (int i = 0; i < 16; ++i) {
      int flat = i * 256 + tid;
      int r = flat >> 5, c = flat & 31;
      As[r * LDSA + c] = (__bf16)X[(size_t)(m0 + r) * D_MODEL + (kt + c)];
    }

    // in-order ASYNCcnt: <=2 retires current tile's 2 copies, leaves the
    // 2 next-tile copies in flight; last iteration drains fully.
    if (it + 1 < NT) wait_asynccnt2(); else wait_asynccnt0();
    __syncthreads();

    const __bf16* Bcur = Bs[it & 1];

    // Fragment loads per CDNA5 16-bit A (16x32) / B (32x16) VGPR layouts:
    //  A: lanes 0-15 M=lane, VGPR v holds K = (v>>2)*16 + half*8 + (v&3)*2, +1
    //  B: lanes 0-15 N=lane, VGPR v holds K = (v>>2)*16 + half*8 + (v&3)*2, +1
    union { v16bf v; unsigned u[8]; } af[4], bf[2];
    #pragma unroll
    for (int mi = 0; mi < 4; ++mi) {
      int m = wm * 64 + mi * 16 + l16;
      #pragma unroll
      for (int v = 0; v < 8; ++v) {
        int kk = ((v >> 2) * 16) + half * 8 + (v & 3) * 2;
        af[mi].u[v] = *(const unsigned*)&As[m * LDSA + kk];
      }
    }
    #pragma unroll
    for (int ni = 0; ni < 2; ++ni) {
      int n = wn * 32 + ni * 16 + l16;
      #pragma unroll
      for (int v = 0; v < 8; ++v) {
        int kk = ((v >> 2) * 16) + half * 8 + (v & 3) * 2;
        bf[ni].u[v] = *(const unsigned*)&Bcur[n * LDSB + kk];
      }
    }
    #pragma unroll
    for (int mi = 0; mi < 4; ++mi)
      #pragma unroll
      for (int ni = 0; ni < 2; ++ni)
        acc[mi][ni] = __builtin_amdgcn_wmma_f32_16x16x32_bf16(
            false, af[mi].v, false, bf[ni].v, (short)0, acc[mi][ni],
            false, false);
  }

  // Epilogue: C/D layout — VGPR j: lanes 0-15 M=j, lanes 16-31 M=j+8; N=l16
  #pragma unroll
  for (int mi = 0; mi < 4; ++mi)
    #pragma unroll
    for (int ni = 0; ni < 2; ++ni) {
      int ng = n0 + wn * 32 + ni * 16 + l16;
      float bv = bias[ng];
      int mbase = m0 + wm * 64 + mi * 16 + half * 8;
      #pragma unroll
      for (int j = 0; j < 8; ++j)
        out[(size_t)(mbase + j) * D_MODEL + ng] = (__bf16)(acc[mi][ni][j] + bv);
    }
}

// ---------------------------------------------------------------------------
__device__ __forceinline__ float wred(float v) {
  #pragma unroll
  for (int off = 16; off > 0; off >>= 1) v += __shfl_xor(v, off, 32);
  return v;
}

// One block per batch. 1024 threads = 32 waves; wave w owns heads w and w+32;
// lane = blade component. gp() done with cross-lane shuffles (ds_bpermute).
__global__ __launch_bounds__(1024) void rotor_scan_kernel(
    const __bf16* __restrict__ rotors,  // [BSZ, SEQ, NHEADS*32] bf16 (unnormalized)
    const float*  __restrict__ Wsync,   // [64,64] f32
    float*        __restrict__ out)     // [BSZ, 64, 32] f32
{
  __shared__ float S[32 * 32];           // Cayley sign table s(a,b), out blade a^b
  __shared__ float revs[32];
  __shared__ float psiL[NHEADS * 33];
  __shared__ float WsL[NHEADS * NHEADS];

  const int tid = threadIdx.x;
  const int k   = tid & 31;
  const int h0  = tid >> 5;            // 0..31
  const int h1  = h0 + 32;
  const int bb  = blockIdx.x;

  { // build Cl(4,1) Cayley sign table (metric = [1,1,1,1,-1])
    int a = tid >> 5, b = tid & 31;
    int sw = 0, t2 = a >> 1;
    while (t2) { sw += __popc(t2 & b); t2 >>= 1; }
    float s = (sw & 1) ? -1.f : 1.f;
    if ((a & b) & 16) s = -s;          // e4e4 = -1
    S[(a << 5) | b] = s;
  }
  if (tid < 32) {
    int g = __popc(tid);
    revs[tid] = (((g * (g - 1)) / 2) & 1) ? -1.f : 1.f;
  }
  #pragma unroll
  for (int i = 0; i < 4; ++i) WsL[tid + i * 1024] = Wsync[tid + i * 1024];
  __syncthreads();

  const float revk = revs[k];
  float p0 = (k == 0) ? 1.f : 0.f;
  float p1 = p0;

  const __bf16* rp = rotors + (size_t)bb * SEQ * (NHEADS * 32) + (h0 << 5) + k;

  for (int s = 0; s < SEQ; ++s) {
    float r0 = (float)rp[0];
    float r1 = (float)rp[1024];        // head h0+32
    __builtin_prefetch(rp + 8 * (NHEADS * 32), 0, 1);   // global_prefetch_b8
    rp += NHEADS * 32;

    // normalize R; inverse scale 1/(|R|^2 + eps) of the normalized rotor
    float n0 = wred(r0 * r0), n1 = wred(r1 * r1);
    float i0 = 1.f / (sqrtf(n0) + EPSV), i1 = 1.f / (sqrtf(n1) + EPSV);
    r0 *= i0; r1 *= i1;
    float invd0 = 1.f / (n0 * i0 * i0 + EPSV);
    float invd1 = 1.f / (n1 * i1 * i1 + EPSV);

    // T = gp(R, psi):  T[k] = sum_a s(a,a^k) R[a] psi[a^k]
    float t0 = 0.f, t1 = 0.f;
    #pragma unroll 4
    for (int a = 0; a < 32; ++a) {
      int j = a ^ k;
      float sg = S[(a << 5) | j];
      t0 = fmaf(sg * __shfl(r0, a, 32), __shfl(p0, j, 32), t0);
      t1 = fmaf(sg * __shfl(r1, a, 32), __shfl(p1, j, 32), t1);
    }
    // U = gp(T, R^-1),  R^-1[j] = revs[j]*R[j]*invd  (fold revs into own lane)
    float rr0 = revk * r0, rr1 = revk * r1;
    float u0 = 0.f, u1 = 0.f;
    #pragma unroll 4
    for (int a = 0; a < 32; ++a) {
      int j = a ^ k;
      float sg = S[(a << 5) | j];
      u0 = fmaf(sg * __shfl(t0, a, 32), __shfl(rr0, j, 32), u0);
      u1 = fmaf(sg * __shfl(t1, a, 32), __shfl(rr1, j, 32), u1);
    }
    u0 *= invd0; u1 *= invd1;
    float m0 = wred(u0 * u0), m1 = wred(u1 * u1);
    p0 = u0 / (sqrtf(m0) + EPSV);
    p1 = u1 / (sqrtf(m1) + EPSV);

    // head sync every 4 steps: psi <- normalize(W_sync @ psi)
    if (((s + 1) & 3) == 0) {
      psiL[h0 * 33 + k] = p0;
      psiL[h1 * 33 + k] = p1;
      __syncthreads();
      float a0 = 0.f, a1 = 0.f;
      #pragma unroll 8
      for (int h = 0; h < NHEADS; ++h) {
        float ph = psiL[h * 33 + k];
        a0 = fmaf(WsL[h0 * 64 + h], ph, a0);
        a1 = fmaf(WsL[h1 * 64 + h], ph, a1);
      }
      __syncthreads();
      float q0 = wred(a0 * a0), q1 = wred(a1 * a1);
      p0 = a0 / (sqrtf(q0) + EPSV);
      p1 = a1 / (sqrtf(q1) + EPSV);
    }
  }

  out[((bb << 6) + h0) * 32 + k] = p0;
  out[((bb << 6) + h1) * 32 + k] = p1;
}

// ---------------------------------------------------------------------------
extern "C" void kernel_launch(void* const* d_in, const int* in_sizes, int n_in,
                              void* d_out, int out_size, void* d_ws, size_t ws_size,
                              hipStream_t stream) {
  const float* X  = (const float*)d_in[0];   // [4,4096,2048]
  const float* Wl = (const float*)d_in[1];   // [2048,2048]
  const float* bl = (const float*)d_in[2];   // [2048]
  const float* Ws = (const float*)d_in[3];   // [64,64]
  float* out = (float*)d_out;                // [4,64,32]

  // workspace: [Wt bf16: 8 MB][rotors bf16: 64 MB]
  __bf16* wt  = (__bf16*)d_ws;
  __bf16* rot = (__bf16*)((char*)d_ws +
                          (size_t)D_MODEL * D_MODEL * sizeof(__bf16));

  cvt_wT_kernel<<<(D_MODEL * D_MODEL) / 256, 256, 0, stream>>>(Wl, wt);

  dim3 grid(D_MODEL / BN, MTOT / BM);        // (16 N, 128 M): N fastest for L2
  lift_gemm_kernel<<<grid, 256, 0, stream>>>(X, wt, bl, rot);

  rotor_scan_kernel<<<BSZ, 1024, 0, stream>>>(rot, Ws, out);
}
